// MLPPool_56195352100976
// MI455X (gfx1250) — matrix operand order repaired and checked by reading, not compiled, not run on hardware
//
#include <hip/hip_runtime.h>

// ---------------------------------------------------------------------------
// Types for CDNA5 WMMA (wave32, 16x16x32 bf16, f32 accumulate)
// ---------------------------------------------------------------------------
typedef __attribute__((ext_vector_type(16))) __bf16 v16bf;
typedef __attribute__((ext_vector_type(8)))  float  v8f;

union Frag16 {
    v16bf        v;
    unsigned int u[8];
    uint4        q[2];
};

__device__ __forceinline__ unsigned int pack2_bf16(float lo, float hi) {
    unsigned int a = __float_as_uint(lo);
    unsigned int b = __float_as_uint(hi);
    a += 0x7FFFu + ((a >> 16) & 1u);   // round-to-nearest-even
    b += 0x7FFFu + ((b >> 16) & 1u);
    return (a >> 16) | (b & 0xFFFF0000u);
}

__device__ __forceinline__ float4 relu4(float4 f) {
    f.x = fmaxf(f.x, 0.f); f.y = fmaxf(f.y, 0.f);
    f.z = fmaxf(f.z, 0.f); f.w = fmaxf(f.w, 0.f);
    return f;
}

// ---------------------------------------------------------------------------
// f32 -> bf16 (RNE) weight conversion
// ---------------------------------------------------------------------------
__global__ void cvt_f32_to_bf16(const float* __restrict__ src,
                                unsigned short* __restrict__ dst, int n) {
    int i = blockIdx.x * blockDim.x + threadIdx.x;
    if (i < n) {
        unsigned int u = __float_as_uint(src[i]);
        u += 0x7FFFu + ((u >> 16) & 1u);
        dst[i] = (unsigned short)(u >> 16);
    }
}

// ---------------------------------------------------------------------------
// WMMA GEMM:  C[M,Nout] = act(A[M,K]) x W[Nout,K]^T   (W pre-converted bf16)
//   block: 256 threads = 8 waves, arranged 4(M) x 2(N)
//   wave tile: 16 rows x 128 cols (8 x v8f accumulators)
//   block tile: 64 rows x 256 cols ; grid = (Nout/256, ceil(M/64))
//   Pipelining: B fragments ping-pong (load b[nt+1] issued before WMMA on
//   b[nt], into the alternate registers -> no WAR hazard nops, one load
//   clause always in flight). A raw data for k+32 is loaded at the top of
//   each k iteration and converted next iteration.
// ---------------------------------------------------------------------------
template <bool A_BF16, bool RELU_A, bool OUT_BF16, bool RELU_OUT>
__global__ __launch_bounds__(256) void gemm_wmma_bf16(
    const void* __restrict__ Ap, const unsigned short* __restrict__ Wb,
    void* __restrict__ Cp, int M, int K, int Nout) {

    const int lane   = threadIdx.x & 31;
    const int wave   = threadIdx.x >> 5;
    const int wm     = wave >> 1;      // 0..3  (M direction)
    const int wn     = wave & 1;       // 0..1  (N direction)
    const int m_lane = lane & 15;
    const int h      = lane >> 4;      // lane half

    const int row_base = blockIdx.y * 64 + wm * 16;
    const int n_base   = blockIdx.x * 256 + wn * 128;

    int arow = row_base + m_lane;
    if (arow > M - 1) arow = M - 1;    // clamp loads; stores are predicated

    const unsigned Ku = (unsigned)K;
    const unsigned a_row_off = (unsigned)arow * Ku;   // elements

    const uint4*  Wq = (const uint4*)Wb;              // uint4 = 8 bf16
    const float4* Af = (const float4*)Ap;
    const uint4*  Aq = (const uint4*)Ap;

    // per-nt B base offsets in uint4 units; advance by k0/8 each step
    unsigned boff[8];
#pragma unroll
    for (int nt = 0; nt < 8; ++nt)
        boff[nt] = (((unsigned)(n_base + nt * 16 + m_lane)) * Ku +
                    (unsigned)(8 * h)) >> 3;

    const unsigned a_base_f = a_row_off >> 2;  // float4 units
    const unsigned a_base_q = a_row_off >> 3;  // uint4 units

    v8f acc[8] = {};

    // ---- prologue: A raw data for k0 = 0
    float4 fA0, fA1, fA2, fA3;
    uint4  qA0, qA1;
    if (A_BF16) {
        qA0 = Aq[a_base_q + h];
        qA1 = Aq[a_base_q + 2 + h];
    } else {
        fA0 = Af[a_base_f + 2 * h];
        fA1 = Af[a_base_f + 2 * h + 1];
        fA2 = Af[a_base_f + 4 + 2 * h];
        fA3 = Af[a_base_f + 5 + 2 * h];
    }

    for (int k0 = 0; k0 < K; k0 += 32) {
        // ---- issue next-iteration A loads early (clamped, branch-free)
        const unsigned kn = (unsigned)((k0 + 32 < K) ? (k0 + 32) : k0);
        float4 fN0, fN1, fN2, fN3;
        uint4  qN0, qN1;
        if (A_BF16) {
            const unsigned ab = a_base_q + (kn >> 3);
            qN0 = Aq[ab + h];
            qN1 = Aq[ab + 2 + h];
        } else {
            const unsigned ab = a_base_f + (kn >> 2);
            fN0 = Af[ab + 2 * h];
            fN1 = Af[ab + 2 * h + 1];
            fN2 = Af[ab + 4 + 2 * h];
            fN3 = Af[ab + 5 + 2 * h];
        }

        // ---- build current A fragment (ISA 16-bit A 16x32 layout)
        Frag16 a;
        if (A_BF16) {
            a.q[0] = qA0;
            a.q[1] = qA1;
        } else {
            float4 f0 = fA0, f1 = fA1, f2 = fA2, f3 = fA3;
            if (RELU_A) { f0 = relu4(f0); f1 = relu4(f1); f2 = relu4(f2); f3 = relu4(f3); }
            a.u[0] = pack2_bf16(f0.x, f0.y);
            a.u[1] = pack2_bf16(f0.z, f0.w);
            a.u[2] = pack2_bf16(f1.x, f1.y);
            a.u[3] = pack2_bf16(f1.z, f1.w);
            a.u[4] = pack2_bf16(f2.x, f2.y);
            a.u[5] = pack2_bf16(f2.z, f2.w);
            a.u[6] = pack2_bf16(f3.x, f3.y);
            a.u[7] = pack2_bf16(f3.z, f3.w);
        }

        // ---- B ping-pong + 8 WMMAs
        const unsigned kq = (unsigned)k0 >> 3;         // uint4 units
        Frag16 bcur, bnxt;
        bcur.q[0] = Wq[boff[0] + kq];
        bcur.q[1] = Wq[boff[0] + kq + 2];
#pragma unroll
        for (int nt = 0; nt < 8; ++nt) {
            if (nt < 7) {
                bnxt.q[0] = Wq[boff[nt + 1] + kq];
                bnxt.q[1] = Wq[boff[nt + 1] + kq + 2];
            }
            acc[nt] = __builtin_amdgcn_wmma_f32_16x16x32_bf16(
                false, a.v, false, bcur.v, (short)0, acc[nt], false, false);
            if (nt < 7) bcur = bnxt;
        }

        // ---- rotate pipelined A data
        if (A_BF16) { qA0 = qN0; qA1 = qN1; }
        else        { fA0 = fN0; fA1 = fN1; fA2 = fN2; fA3 = fN3; }
    }

    // ---- store: C/D layout: col = n_base + nt*16 + (lane&15), row = r + 8h
    const bool full_tile = (row_base + 15) < M;
#pragma unroll
    for (int nt = 0; nt < 8; ++nt) {
        const unsigned col = (unsigned)(n_base + nt * 16 + m_lane);
#pragma unroll
        for (int r = 0; r < 8; ++r) {
            const int ro = row_base + r + 8 * h;
            if (full_tile || ro < M) {
                float vv = acc[nt][r];
                if (RELU_OUT) vv = fmaxf(vv, 0.f);
                if (OUT_BF16) {
                    unsigned int u = __float_as_uint(vv);
                    u += 0x7FFFu + ((u >> 16) & 1u);
                    ((unsigned short*)Cp)[(size_t)ro * Nout + col] =
                        (unsigned short)(u >> 16);
                } else {
                    ((float*)Cp)[(size_t)ro * Nout + col] = vv;
                }
            }
        }
    }
}

// ---------------------------------------------------------------------------
// COO SpMM (segment-sum): one wave per edge, lane covers 16 of 512 columns.
// Gathers bf16 source row, scales by edge_val, f32 atomic-adds into dest row.
// ---------------------------------------------------------------------------
__global__ __launch_bounds__(256) void spmm_scatter_bf16(
    const unsigned short* __restrict__ hsrc, const int* __restrict__ erow,
    const int* __restrict__ ecol, const float* __restrict__ ev,
    float* __restrict__ hdst, int E) {

    const int e = blockIdx.x * (blockDim.x >> 5) + (threadIdx.x >> 5);
    if (e >= E) return;
    const int lane = threadIdx.x & 31;

    const int r = erow[e];
    const int c = ecol[e];
    const float v = ev[e];

    const uint4* src = (const uint4*)(hsrc + (size_t)c * 512) + lane * 2; // 16 bf16
    float* dst = hdst + (size_t)r * 512 + lane * 16;

    uint4 q0 = src[0];
    uint4 q1 = src[1];
    unsigned int w[8] = {q0.x, q0.y, q0.z, q0.w, q1.x, q1.y, q1.z, q1.w};
#pragma unroll
    for (int i = 0; i < 8; ++i) {
        float lo = __uint_as_float((w[i] & 0x0000FFFFu) << 16);
        float hi = __uint_as_float(w[i] & 0xFFFF0000u);
        __hip_atomic_fetch_add(dst + 2 * i + 0, v * lo, __ATOMIC_RELAXED,
                               __HIP_MEMORY_SCOPE_AGENT);
        __hip_atomic_fetch_add(dst + 2 * i + 1, v * hi, __ATOMIC_RELAXED,
                               __HIP_MEMORY_SCOPE_AGENT);
    }
}

// ---------------------------------------------------------------------------
// Launch: x -> GEMM0(bf16 out) -> SpMM(f32 acc) -> GEMM1(relu in+out, bf16 out)
//         -> GEMM2(f32 out)
// ---------------------------------------------------------------------------
extern "C" void kernel_launch(void* const* d_in, const int* in_sizes, int n_in,
                              void* d_out, int out_size, void* d_ws, size_t ws_size,
                              hipStream_t stream) {
    const float* x   = (const float*)d_in[0];
    const int*  erow = (const int*)d_in[1];
    const int*  ecol = (const int*)d_in[2];
    const float* ev  = (const float*)d_in[3];
    const float* W0  = (const float*)d_in[4];
    const float* W1  = (const float*)d_in[5];
    const float* W2  = (const float*)d_in[6];

    const int DIN = 512, DH = 512, DOUT = 256;
    const int N = in_sizes[0] / DIN;     // 100000 nodes
    const int E = in_sizes[1];           // 3.2M edges

    // workspace layout: [h1 f32 N*DH][h0/h2 bf16 N*DH][W0b][W1b][W2b]
    char* ws = (char*)d_ws;
    float* h1 = (float*)ws;
    unsigned short* h0b = (unsigned short*)(ws + (size_t)N * DH * sizeof(float));
    unsigned short* W0b = h0b + (size_t)N * DH;
    unsigned short* W1b = W0b + (size_t)DH * DIN;
    unsigned short* W2b = W1b + (size_t)DH * DH;

    // Weights -> bf16 (tiny, L2-resident afterwards)
    cvt_f32_to_bf16<<<(DH * DIN + 255) / 256, 256, 0, stream>>>(W0, W0b, DH * DIN);
    cvt_f32_to_bf16<<<(DH * DH  + 255) / 256, 256, 0, stream>>>(W1, W1b, DH * DH);
    cvt_f32_to_bf16<<<(DOUT * DH + 255) / 256, 256, 0, stream>>>(W2, W2b, DOUT * DH);

    const int gy = (N + 63) / 64;

    // L0: h0 = x @ W0^T   (f32 A, bf16 out -> halves SpMM gather traffic)
    gemm_wmma_bf16<false, false, true, false>
        <<<dim3(DH / 256, gy), 256, 0, stream>>>(x, W0b, h0b, N, DIN, DH);

    // SpMM: h1 = scatter-add( edge_val * h0[edge_col] ) into edge_row
    hipMemsetAsync(h1, 0, (size_t)N * DH * sizeof(float), stream);
    spmm_scatter_bf16<<<(E + 7) / 8, 256, 0, stream>>>(h0b, erow, ecol, ev, h1, E);

    // L1: h2 = relu( relu(h1) @ W1^T ), bf16 out (reuses h0 buffer)
    gemm_wmma_bf16<false, true, true, true>
        <<<dim3(DH / 256, gy), 256, 0, stream>>>(h1, W1b, h0b, N, DH, DH);

    // L2: out = h2 @ W2^T, f32 out
    gemm_wmma_bf16<true, false, false, false>
        <<<dim3(DOUT / 256, gy), 256, 0, stream>>>(h0b, W2b, d_out, N, DH, DOUT);
}
